// StyleGeneratorBlock_45071386804779
// MI455X (gfx1250) — matrix operand order repaired and checked by reading, not compiled, not run on hardware
//
#include <hip/hip_runtime.h>

// ---------------------------------------------------------------------------
// Types for CDNA5 WMMA (gfx1250, wave32)
// ---------------------------------------------------------------------------
typedef __bf16 bf16;
typedef bf16  v16bf __attribute__((ext_vector_type(16)));
typedef bf16  v8bf  __attribute__((ext_vector_type(8)));
typedef float v8f   __attribute__((ext_vector_type(8)));

union Frag16 {
    v16bf v;
    v8bf  h[2];
};

// Problem constants (reference: B=8, C=L=512, H=W=64, 3x3 conv)
#define BATCH   8
#define CH      512
#define LAT     512
#define HH      64
#define WW      64
#define PH      66          // padded spatial (zero halo)
#define KC      32          // K chunk (ci per WMMA group)
#define NKC     (CH / KC)   // 16 k-chunks
#define CO_BLK  128         // Co per workgroup
#define ROWS    2           // output rows per workgroup
#define TROWS   4           // input rows staged in LDS (ROWS + 2 halo)
#define TILE_E  (TROWS * PH * KC)      // elements per LDS tile buffer
#define NCHUNK  (TROWS * PH * 4)       // 16B async chunks per tile
#define SCALE_EQ 0.014731391274719738f // 1/sqrt(512*9)
#define LRELU   0.2f
#define ACTG    1.4142135623730951f

static __device__ __forceinline__ void async_b128_to_lds(unsigned lds_addr,
                                                         const bf16* gptr) {
    unsigned long long ga = (unsigned long long)(uintptr_t)gptr;
    asm volatile("global_load_async_to_lds_b128 %0, %1, off"
                 :: "v"(lds_addr), "v"(ga) : "memory");
}
static __device__ __forceinline__ void wait_asynccnt0() {
#if __has_builtin(__builtin_amdgcn_s_wait_asynccnt)
    __builtin_amdgcn_s_wait_asynccnt(0);
#else
    asm volatile("s_wait_asynccnt 0x0" ::: "memory");
#endif
}

// ---------------------------------------------------------------------------
// Kernel 0a: x (f32 NCHW) -> xT (bf16, padded NHWC [b][66][66][512], zero halo)
// grid = (66, 8), block = 256
// ---------------------------------------------------------------------------
__global__ __launch_bounds__(256) void transpose_pad_kernel(
    const float* __restrict__ x, bf16* __restrict__ xT)
{
    const int py = blockIdx.x;           // 0..65
    const int b  = blockIdx.y;
    const int tid = threadIdx.x;
    const int y = py - 1;
    const bool yin = (unsigned)y < (unsigned)HH;
    bf16* row = xT + (((size_t)b * PH + py) * PH) * CH;
    for (int px = 0; px < PH; ++px) {
        const int xx = px - 1;
        const bool in = yin && ((unsigned)xx < (unsigned)WW);
        for (int ci = tid; ci < CH; ci += 256) {
            float v = in ? x[(((size_t)b * CH + ci) * HH + y) * WW + xx] : 0.0f;
            row[(size_t)px * CH + ci] = (bf16)v;
        }
    }
}

// ---------------------------------------------------------------------------
// Kernel 0b: zero a bf16 buffer (for y1T halo; 16B per thread-iter)
// ---------------------------------------------------------------------------
__global__ __launch_bounds__(256) void zero_kernel(uint4* __restrict__ p, int n16) {
    int i = blockIdx.x * 256 + threadIdx.x;
    if (i < n16) p[i] = make_uint4(0, 0, 0, 0);
}

// ---------------------------------------------------------------------------
// Kernel 1: style[b, ci] = w[b,:] . mod_w[ci,:] + mod_b[ci]
// ---------------------------------------------------------------------------
__global__ __launch_bounds__(256) void style_kernel(
    const float* __restrict__ w, const float* __restrict__ mod_w,
    const float* __restrict__ mod_b, float* __restrict__ style)
{
    int idx = blockIdx.x * 256 + threadIdx.x;
    int b  = idx >> 9;
    int ci = idx & (CH - 1);
    const float* wr = w + (size_t)b * LAT;
    const float* mr = mod_w + (size_t)ci * LAT;
    float acc = 0.0f;
#pragma unroll 4
    for (int l = 0; l < LAT; ++l) acc += wr[l] * mr[l];
    style[idx] = acc + mod_b[ci];
}

// ---------------------------------------------------------------------------
// Kernel 2: demod + write bf16 modulated weights wm[b][tap][co][ci]
// ---------------------------------------------------------------------------
__global__ __launch_bounds__(256) void modw_kernel(
    const float* __restrict__ style, const float* __restrict__ weight,
    bf16* __restrict__ wm)
{
    int blk = blockIdx.x;
    int b  = blk >> 9;
    int co = blk & (CH - 1);
    int tid = threadIdx.x;

    __shared__ float red[256];
    const float* st = style + (size_t)b * CH;

    float partial = 0.0f;
    for (int ci = tid; ci < CH; ci += 256) {
        float s = st[ci];
        const float* wp = weight + ((size_t)co * CH + ci) * 9;
        float wsq = 0.0f;
#pragma unroll
        for (int t = 0; t < 9; ++t) wsq += wp[t] * wp[t];
        partial += s * s * wsq;
    }
    red[tid] = partial;
    __syncthreads();
    for (int s = 128; s > 0; s >>= 1) {
        if (tid < s) red[tid] += red[tid + s];
        __syncthreads();
    }
    float demod = rsqrtf(SCALE_EQ * SCALE_EQ * red[0] + 1e-8f);
    float dm = SCALE_EQ * demod;

    for (int ci = tid; ci < CH; ci += 256) {
        float s = st[ci] * dm;
        const float* wp = weight + ((size_t)co * CH + ci) * 9;
#pragma unroll
        for (int t = 0; t < 9; ++t) {
            size_t o = ((((size_t)b * 9 + t) * CH) + co) * CH + ci;
            wm[o] = (bf16)(wp[t] * s);
        }
    }
}

// ---------------------------------------------------------------------------
// Kernel 3: modulated 3x3 conv via WMMA bf16 on padded NHWC bf16 input.
//   - double-buffered LDS tiles filled with global_load_async_to_lds_b128
//     (ASYNCcnt), prefetching chunk k+1 while computing chunk k
//   - 9 shifted GEMMs, 2 output rows, fused noise/bias/leaky-relu epilogue
//   - writes bf16 padded NHWC (intermediate) or f32 NCHW (final)
// grid = (H/2=32, Co/128=4, B=8), block = 256 (8 waves)
// ---------------------------------------------------------------------------
__global__ __launch_bounds__(256) void conv_kernel(
    const bf16*  __restrict__ inT,     // [B][66][66][512] bf16 (zero halo)
    const bf16*  __restrict__ wm,      // [B][9][Co][Ci]   bf16
    const float* __restrict__ noise,   // [B,1,H,W]
    const float* __restrict__ gain_p,  // scalar
    const float* __restrict__ bias,    // [C]
    bf16*  __restrict__ outT,          // [B][66][66][512] bf16 (intermediate)
    float* __restrict__ outF,          // [B,C,H,W] f32 (final)
    int final_f32)
{
    const int y0      = blockIdx.x * ROWS;       // first output row
    const int co_blk  = blockIdx.y * CO_BLK;
    const int b       = blockIdx.z;
    const int tid     = threadIdx.x;
    const int wave    = tid >> 5;
    const int lane    = tid & 31;
    const int half    = lane >> 4;
    const int ln      = lane & 15;
    const int co_base = co_blk + wave * 16;

    // Ping-pong LDS tiles: [t=0..3][px=0..65][ci=0..31], ci innermost (bf16)
    __shared__ __align__(16) bf16 tile[2][TILE_E];     // 2 x 16896 B
    const unsigned ldsb[2] = { (unsigned)(uintptr_t)(&tile[0][0]),
                               (unsigned)(uintptr_t)(&tile[1][0]) };

    const bf16* in_b = inT + ((size_t)b * PH + y0) * PH * CH;  // row py = y0+t

    // issue async copies for k-chunk kc into LDS buffer `base`
    auto stage = [&](int kc, unsigned base) {
        const int ci0 = kc * KC;
        for (int c = tid; c < NCHUNK; c += 256) {
            int cic  = c & 3;            // 16B chunk within 64B ci-run
            int rest = c >> 2;           // t*66 + px
            int px   = rest % PH;
            int t    = rest / PH;
            const bf16* gp = in_b + (((size_t)t * PH + px) * CH) + ci0 + cic * 8;
            async_b128_to_lds(base + (unsigned)c * 16, gp);
        }
    };

    v8f acc[ROWS][4];
#pragma unroll
    for (int r = 0; r < ROWS; ++r)
#pragma unroll
        for (int i = 0; i < 4; ++i) acc[r][i] = (v8f){0,0,0,0,0,0,0,0};

    // Prologue: fill buffer 0
    stage(0, ldsb[0]);
    wait_asynccnt0();
    __syncthreads();

    for (int kc = 0; kc < NKC; ++kc) {
        const int cur = kc & 1;
        const int ci0 = kc * KC;
        const bf16* tcur = tile[cur];

        // Prefetch next k-chunk into the other buffer (safe: that buffer was
        // last read in iteration kc-1, all waves passed the barrier since)
        if (kc + 1 < NKC) stage(kc + 1, ldsb[cur ^ 1]);

        // ---- 9 shifted GEMM accumulations over 2 output rows ----
#pragma unroll
        for (int tap = 0; tap < 9; ++tap) {
            const int dy = tap / 3;
            const int dx = tap % 3;

            // A fragment: wm[b][tap][co_base+ln][ci0 + K]
            Frag16 a;
            const bf16* ap = wm + ((((size_t)b * 9 + tap) * CH) + (co_base + ln)) * CH
                                + ci0 + half * 8;
            a.h[0] = *(const v8bf*)(ap);
            a.h[1] = *(const v8bf*)(ap + 16);

#pragma unroll
            for (int r = 0; r < ROWS; ++r) {
#pragma unroll
                for (int nt = 0; nt < 4; ++nt) {
                    const int col = nt * 16 + ln + dx;    // stored col = x + 1
                    Frag16 bfr;
                    const bf16* bp = &tcur[((dy + r) * PH + col) * KC + half * 8];
                    bfr.h[0] = *(const v8bf*)(bp);
                    bfr.h[1] = *(const v8bf*)(bp + 16);
                    acc[r][nt] = __builtin_amdgcn_wmma_f32_16x16x32_bf16(
                        false, a.v, false, bfr.v, (short)0, acc[r][nt], false, false);
                }
            }
        }

        // Next tile has had the whole compute phase to land; cheap wait.
        wait_asynccnt0();
        __syncthreads();
    }

    // ---- Fused epilogue: + gain*noise + bias, leaky relu * sqrt(2) ----
    const float gain = gain_p[0];
#pragma unroll
    for (int r = 0; r < ROWS; ++r) {
        const int yr = y0 + r;
#pragma unroll
        for (int nt = 0; nt < 4; ++nt) {
            const int x = nt * 16 + ln;
            const float nz = gain * noise[((size_t)b * HH + yr) * WW + x];
            float vals[8];
#pragma unroll
            for (int v = 0; v < 8; ++v) {
                const int co = co_base + half * 8 + v;   // C/D layout: M = 8*half + v
                float val = acc[r][nt][v] + nz + bias[co];
                vals[v] = (val >= 0.0f ? val : LRELU * val) * ACTG;
            }
            if (final_f32) {
#pragma unroll
                for (int v = 0; v < 8; ++v) {
                    const int co = co_base + half * 8 + v;
                    outF[(((size_t)b * CH + co) * HH + yr) * WW + x] = vals[v];
                }
            } else {
                v8bf o;
#pragma unroll
                for (int v = 0; v < 8; ++v) o[v] = (bf16)vals[v];
                bf16* op = outT + ((((size_t)b * PH + (yr + 1)) * PH + (x + 1)) * CH)
                                + co_base + half * 8;
                *(v8bf*)op = o;
            }
        }
    }
}

// ---------------------------------------------------------------------------
// Host-side orchestration
// ---------------------------------------------------------------------------
extern "C" void kernel_launch(void* const* d_in, const int* in_sizes, int n_in,
                              void* d_out, int out_size, void* d_ws, size_t ws_size,
                              hipStream_t stream) {
    const float* x       = (const float*)d_in[0];
    const float* w1      = (const float*)d_in[1];
    const float* w2      = (const float*)d_in[2];
    const float* noise1  = (const float*)d_in[3];
    const float* noise2  = (const float*)d_in[4];
    const float* mod1_w  = (const float*)d_in[5];
    const float* mod1_b  = (const float*)d_in[6];
    const float* conv1_w = (const float*)d_in[7];
    const float* gain1   = (const float*)d_in[8];
    const float* bias1   = (const float*)d_in[9];
    const float* mod2_w  = (const float*)d_in[10];
    const float* mod2_b  = (const float*)d_in[11];
    const float* conv2_w = (const float*)d_in[12];
    const float* gain2   = (const float*)d_in[13];
    const float* bias2   = (const float*)d_in[14];
    float* out = (float*)d_out;

    // Workspace layout
    const size_t PAD_BYTES = (size_t)BATCH * PH * PH * CH * sizeof(bf16);   // 35,684,352
    const size_t WM_BYTES  = (size_t)BATCH * 9 * CH * CH * sizeof(bf16);    // 37,748,736
    char* ws = (char*)d_ws;
    bf16*  xT    = (bf16*)ws;
    bf16*  y1T   = (bf16*)(ws + PAD_BYTES);
    bf16*  wmod  = (bf16*)(ws + 2 * PAD_BYTES);
    float* style = (float*)(ws + 2 * PAD_BYTES + WM_BYTES);

    const dim3 convGrid(HH / ROWS, CH / CO_BLK, BATCH);   // (32, 4, 8)
    const int n16 = (int)(PAD_BYTES / 16);

    // ---- Prep: convert+pad x; zero y1T (halo must be 0 for layer 2) ----
    transpose_pad_kernel<<<dim3(PH, BATCH), 256, 0, stream>>>(x, xT);
    zero_kernel<<<(n16 + 255) / 256, 256, 0, stream>>>((uint4*)y1T, n16);

    // ---- Layer 1 ----
    style_kernel<<<(BATCH * CH) / 256, 256, 0, stream>>>(w1, mod1_w, mod1_b, style);
    modw_kernel<<<BATCH * CH, 256, 0, stream>>>(style, conv1_w, wmod);
    conv_kernel<<<convGrid, 256, 0, stream>>>(xT, wmod, noise1, gain1, bias1,
                                              y1T, out, 0);

    // ---- Layer 2 ----
    style_kernel<<<(BATCH * CH) / 256, 256, 0, stream>>>(w2, mod2_w, mod2_b, style);
    modw_kernel<<<BATCH * CH, 256, 0, stream>>>(style, conv2_w, wmod);
    conv_kernel<<<convGrid, 256, 0, stream>>>(y1T, wmod, noise2, gain2, bias2,
                                              y1T, out, 1);
}
